// DiagnosticRNN_48026324304699
// MI455X (gfx1250) — matrix operand order repaired and checked by reading, not compiled, not run on hardware
//
#include <hip/hip_runtime.h>

typedef __attribute__((ext_vector_type(16))) _Float16 v16h;
typedef __attribute__((ext_vector_type(2)))  __fp16   v2hf;  // return type of cvt_pkrtz
typedef __attribute__((ext_vector_type(8)))  float    v8f;
typedef __attribute__((ext_vector_type(4)))  float    v4f;

#define BATCH 4096
#define SEQT  1024
#define HID   64
#define NCLS  10
#define RB    32      // batch rows per block (2 row-tiles of 16)
#define TC    64      // timestep chunk staged in LDS
#define HS    72      // padded LDS row stride for h (halves) -> conflict-free b128 A loads
#define XRS   36      // padded LDS row stride for transposed x (floats, 16B aligned)

// Hidden-dimension permutation: LDS position p <-> original hidden index.
// pos = 32g + 2l + j for orig = 32g + 16j + l. This makes each lane's two
// per-row tanh results adjacent in LDS (single packed b32 store) while A-operand
// loads stay contiguous. Whh rows are permuted to match when building B operands.
static __device__ __forceinline__ int posToOrig(int p) {
    return (p & 32) + ((p & 1) << 4) + ((p & 30) >> 1);
}
static __device__ __forceinline__ int origToPos(int k) {
    return (k & 32) + ((k & 15) << 1) + ((k >> 4) & 1);
}

// Single-instruction tanh on CDNA5 (V_TANH_F32 hardware TRANS op).
static __device__ __forceinline__ float fast_tanh(float v) {
#if __has_builtin(__builtin_amdgcn_tanhf)
    return __builtin_amdgcn_tanhf(v);
#else
    float r;
    asm("v_tanh_f32 %0, %1" : "=v"(r) : "v"(v));
    return r;
#endif
}

__global__ __launch_bounds__(128)
void rnn_wmma_kernel(const float* __restrict__ x,
                     const float* __restrict__ Whx,
                     const float* __restrict__ Whh,
                     const float* __restrict__ Wph,
                     const float* __restrict__ bh,
                     const float* __restrict__ bp,
                     float* __restrict__ out)
{
    __shared__ __align__(16) _Float16 hbuf[2][RB * HS];  // ping-pong hidden state (f16, permuted cols)
    __shared__ __align__(16) float    xbuf[TC * XRS];    // staged x chunk, transposed [t][row]

    const int tid   = threadIdx.x;
    const int lane  = tid & 31;
    const int wave  = tid >> 5;
    const int r0    = (wave & 1) * 16;   // 16-row batch tile within block
    const int n0    = (wave >> 1) * 32;  // 32-col hidden half
    const int gb0   = blockIdx.x * RB;   // first global batch row of block

    const int lx = lane & 15;            // column-in-tile / row-in-tile index
    const int lh = lane >> 4;            // half-wave select

    // ---- Whh resident in B-operand layout, rows gathered through the permutation ----
    // B tile (Ntile j, Kchunk c): half i holds Whh[posToOrig(c*32 + lh*16 + i)][n0 + j*16 + lx]
    v16h b00, b01, b10, b11;
    {
        const int colA = n0 + lx;
        const int colB = n0 + 16 + lx;
        const int kb0  = lh * 16;
#pragma unroll
        for (int i = 0; i < 16; ++i) {
            b00[i] = (_Float16)Whh[posToOrig(kb0 + i) * HID + colA];       // j=0, K-pos 0..31
            b01[i] = (_Float16)Whh[posToOrig(32 + kb0 + i) * HID + colA];  // j=0, K-pos 32..63
            b10[i] = (_Float16)Whh[posToOrig(kb0 + i) * HID + colB];       // j=1, K-pos 0..31
            b11[i] = (_Float16)Whh[posToOrig(32 + kb0 + i) * HID + colB];  // j=1, K-pos 32..63
        }
    }
    const float whx0 = Whx[n0 + lx];
    const float whx1 = Whx[n0 + 16 + lx];
    const float bh0  = bh[n0 + lx];
    const float bh1  = bh[n0 + 16 + lx];

    // h0 = 0
    for (int i = tid; i < RB * HS; i += 128) hbuf[0][i] = (_Float16)0.0f;

    const int rowbase = r0 + lh * 8;       // first of the 8 C/D rows this lane holds
    const int stpos   = n0 + 2 * lx;       // packed-pair LDS column position for stores
    int cur = 0;

    for (int tc0 = 0; tc0 < SEQT; tc0 += TC) {
        __syncthreads();  // previous chunk's x reads complete before overwrite
        // stage x[gb0..gb0+RB-1][tc0..tc0+TC-1] transposed (coalesced global reads)
#pragma unroll
        for (int k = 0; k < (RB * TC) / 128; ++k) {
            int e   = tid + k * 128;
            int row = e >> 6;          // TC == 64
            int tt  = e & (TC - 1);
            xbuf[tt * XRS + row] = x[(size_t)(gb0 + row) * SEQT + tc0 + tt];
        }
        if (tc0 + TC < SEQT)  // CDNA5 global_prefetch_b8 for next chunk
            __builtin_prefetch(&x[(size_t)(gb0 + (tid & (RB - 1))) * SEQT + tc0 + TC], 0, 1);
        __syncthreads();

#pragma unroll 2
        for (int tt = 0; tt < TC; ++tt) {
            // ---- A operand: h_{t-1} rows r0..r0+15, K-pos 0..63 (16-bit A layout) ----
            const _Float16* hc = &hbuf[cur][0];
            const int abase = (r0 + lx) * HS + lh * 16;
            v16h a0, a1;
#pragma unroll
            for (int i = 0; i < 8; ++i) {
                a0[i]     = hc[abase + i];        // K-pos = lh*16 + i
                a0[i + 8] = hc[abase + 8 + i];    // K-pos = lh*16 + 8 + i
                a1[i]     = hc[abase + 32 + i];   // K-pos = 32 + lh*16 + i
                a1[i + 8] = hc[abase + 40 + i];   // K-pos = 32 + lh*16 + 8 + i
            }
            // ---- per-row x_t: 8 contiguous floats, two aligned b128 broadcasts ----
            const float* xr = &xbuf[tt * XRS + rowbase];
            v4f xlo = *(const v4f*)(xr);
            v4f xhi = *(const v4f*)(xr + 4);

            // ---- C init = x_t*Whx + bh, then 4 WMMAs (2 N-tiles x 2 K-chunks) ----
            v8f acc0, acc1;
#pragma unroll
            for (int v = 0; v < 4; ++v) {
                acc0[v]     = __builtin_fmaf(xlo[v], whx0, bh0);
                acc0[v + 4] = __builtin_fmaf(xhi[v], whx0, bh0);
                acc1[v]     = __builtin_fmaf(xlo[v], whx1, bh1);
                acc1[v + 4] = __builtin_fmaf(xhi[v], whx1, bh1);
            }
            acc0 = __builtin_amdgcn_wmma_f32_16x16x32_f16(false, a0, false, b00, (short)0, acc0, false, false);
            acc0 = __builtin_amdgcn_wmma_f32_16x16x32_f16(false, a1, false, b01, (short)0, acc0, false, false);
            acc1 = __builtin_amdgcn_wmma_f32_16x16x32_f16(false, a0, false, b10, (short)0, acc1, false, false);
            acc1 = __builtin_amdgcn_wmma_f32_16x16x32_f16(false, a1, false, b11, (short)0, acc1, false, false);

            // ---- h_t = tanh(acc): packed cvt + single b32 store per row ----
            _Float16* hn = &hbuf[cur ^ 1][0];
#pragma unroll
            for (int v = 0; v < 8; ++v) {
                v2hf pr = __builtin_amdgcn_cvt_pkrtz(fast_tanh(acc0[v]), fast_tanh(acc1[v]));
                *(v2hf*)(&hn[(rowbase + v) * HS + stpos]) = pr;
            }
            cur ^= 1;
            __syncthreads();
        }
    }

    // ---- final projection: p = h_T @ Wph + bp (un-permute hidden index) ----
    const _Float16* hf = &hbuf[cur][0];
    for (int o = tid; o < RB * NCLS; o += 128) {
        int row = o / NCLS;
        int cls = o - row * NCLS;
        float s = bp[cls];
#pragma unroll
        for (int k = 0; k < HID; ++k)
            s = __builtin_fmaf((float)hf[row * HS + origToPos(k)], Wph[k * NCLS + cls], s);
        out[(size_t)(gb0 + row) * NCLS + cls] = s;
    }
}

extern "C" void kernel_launch(void* const* d_in, const int* in_sizes, int n_in,
                              void* d_out, int out_size, void* d_ws, size_t ws_size,
                              hipStream_t stream) {
    const float* x   = (const float*)d_in[0];
    const float* Whx = (const float*)d_in[1];
    const float* Whh = (const float*)d_in[2];
    const float* Wph = (const float*)d_in[3];
    const float* bh  = (const float*)d_in[4];
    const float* bp  = (const float*)d_in[5];
    float* out = (float*)d_out;

    dim3 grid(BATCH / RB);   // 128 blocks x 4 waves = 512 waves; each block's 4 waves share one 32-row recurrence
    dim3 block(128);
    hipLaunchKernelGGL(rnn_wmma_kernel, grid, block, 0, stream,
                       x, Whx, Whh, Wph, bh, bp, out);
}
